// InputFwdHadamardWrapper_39986145526404
// MI455X (gfx1250) — matrix operand order repaired and checked by reading, not compiled, not run on hardware
//
#include <hip/hip_runtime.h>

// ---------------------------------------------------------------------------
// Problem constants (match reference): B=4, S=2048, D_IN=D_OUT=4096.
// M = B*S = 8192 rows. FWHT over last axis then GEMM with W^T plus bias.
// ---------------------------------------------------------------------------

typedef __attribute__((ext_vector_type(16))) __bf16 v16bf;
typedef __attribute__((ext_vector_type(8)))  __bf16 v8bf;
typedef __attribute__((ext_vector_type(8)))  float  v8f;

#define D_DIM   4096
#define TILE_M  128
#define TILE_N  128
#define TILE_K  32
#define LDT     40          // 32 + 8 halfs pad => 80B row stride (16B aligned)
#define TILE_HALFS (TILE_M * LDT)

// ----- bf16 helpers (round-to-nearest-even split) ---------------------------
__device__ __forceinline__ unsigned short f32_to_bf16(float f) {
    unsigned u = __float_as_uint(f);
    unsigned r = u + 0x7FFFu + ((u >> 16) & 1u);
    return (unsigned short)(r >> 16);
}
__device__ __forceinline__ float bf16_to_f32(unsigned short h) {
    return __uint_as_float(((unsigned)h) << 16);
}

// ----- ASYNCcnt wait ---------------------------------------------------------
#if __has_builtin(__builtin_amdgcn_s_wait_asynccnt)
#define WAIT_ASYNC(n) __builtin_amdgcn_s_wait_asynccnt(n)
#else
#define WAIT_ASYNC(n) asm volatile("s_wait_asynccnt %0" :: "i"(n) : "memory")
#endif

// low 32 bits of a generic (flat) LDS address == LDS byte offset
__device__ __forceinline__ unsigned lds_off(const void* p) {
    return (unsigned)(unsigned long long)p;
}

// ---------------------------------------------------------------------------
// Kernel 1: per-row FWHT (n=4096) in LDS, scale by 1/64, split to bf16 hi/lo.
// One block (256 threads) per row.
// ---------------------------------------------------------------------------
__global__ void fwht_split_kernel(const float* __restrict__ x,
                                  unsigned short* __restrict__ xhi,
                                  unsigned short* __restrict__ xlo) {
    __shared__ __align__(16) float s[D_DIM];
    const int t = threadIdx.x;                 // 0..255
    const size_t row = blockIdx.x;
    const float* src = x + row * (size_t)D_DIM;

    #pragma unroll
    for (int i = 0; i < 4; ++i) {
        int idx = t + i * 256;
        ((float4*)s)[idx] = ((const float4*)src)[idx];
    }
    __syncthreads();

    for (int h = 1; h < D_DIM; h <<= 1) {
        #pragma unroll
        for (int j = 0; j < 8; ++j) {
            int p = t + j * 256;                          // pair id 0..2047
            int i = ((p & ~(h - 1)) << 1) | (p & (h - 1));
            float a = s[i];
            float b = s[i + h];
            s[i]     = a + b;
            s[i + h] = a - b;
        }
        __syncthreads();
    }

    const float scale = 1.0f / 64.0f;   // 1/sqrt(4096)
    #pragma unroll
    for (int i = 0; i < 16; ++i) {
        int idx = t + i * 256;
        float v = s[idx] * scale;
        unsigned short hi = f32_to_bf16(v);
        unsigned short lo = f32_to_bf16(v - bf16_to_f32(hi));
        xhi[row * D_DIM + idx] = hi;
        xlo[row * D_DIM + idx] = lo;
    }
}

// ---------------------------------------------------------------------------
// Kernel 2: weight f32 -> bf16 hi/lo split (elementwise, 4 per thread).
// ---------------------------------------------------------------------------
__global__ void weight_split_kernel(const float* __restrict__ w,
                                    unsigned short* __restrict__ whi,
                                    unsigned short* __restrict__ wlo) {
    size_t i = ((size_t)blockIdx.x * blockDim.x + threadIdx.x) * 4;
    float4 v = *(const float4*)(w + i);
    ushort4 hv, lv;
    hv.x = f32_to_bf16(v.x); lv.x = f32_to_bf16(v.x - bf16_to_f32(hv.x));
    hv.y = f32_to_bf16(v.y); lv.y = f32_to_bf16(v.y - bf16_to_f32(hv.y));
    hv.z = f32_to_bf16(v.z); lv.z = f32_to_bf16(v.z - bf16_to_f32(hv.z));
    hv.w = f32_to_bf16(v.w); lv.w = f32_to_bf16(v.w - bf16_to_f32(hv.w));
    *(ushort4*)(whi + i) = hv;
    *(ushort4*)(wlo + i) = lv;
}

// ---------------------------------------------------------------------------
// WMMA fragment loaders (wave32 layouts per CDNA5 ISA 7.12.2).
// A (16x32, 16-bit): lane half h = lane>>4, M = lane&15.
//   elements 0..7  = K  h*8   .. h*8+7      (16B chunk)
//   elements 8..15 = K 16+h*8 .. 16+h*8+7   (16B chunk)
// B (32x16, 16-bit): N = lane&15, elements 0..7 = K h*8 .. h*8+7, 8..15 = K 16+h*8 .. 16+h*8+7.
// C/D (16x16 f32): VGPR r -> M = r + 8*h, N = lane&15.
// ---------------------------------------------------------------------------
__device__ __forceinline__ v16bf load_a_frag(const __bf16* __restrict__ tile) {
    const int lane = threadIdx.x & 31;
    const int m = lane & 15;
    const int h = lane >> 4;
    const __bf16* r = tile + m * LDT;
    v8bf x0 = *(const v8bf*)(r + h * 8);
    v8bf x1 = *(const v8bf*)(r + 16 + h * 8);
    v16bf out;
    #pragma unroll
    for (int i = 0; i < 8; ++i) { out[i] = x0[i]; out[i + 8] = x1[i]; }
    return out;
}

__device__ __forceinline__ v16bf load_b_frag(const __bf16* __restrict__ tile) {
    const int lane = threadIdx.x & 31;
    const int n = lane & 15;
    const int h = lane >> 4;
    const __bf16* r = tile + n * LDT + h * 8;
    v8bf x0 = *(const v8bf*)(r);
    v8bf x1 = *(const v8bf*)(r + 16);
    v16bf out;
    #pragma unroll
    for (int i = 0; i < 8; ++i) { out[i] = x0[i]; out[i + 8] = x1[i]; }
    return out;
}

// async global->LDS copy of one 128x32 bf16 tile (512 16B chunks, 2/thread)
// GLOBAL_LOAD_ASYNC_TO_LDS_B128: VDST = per-lane LDS byte address,
// VADDR = per-lane 64-bit global address, no SADDR. Tracked by ASYNCcnt.
__device__ __forceinline__ void async_tile_load(const unsigned short* __restrict__ g,
                                                unsigned lbase, int t) {
    #pragma unroll
    for (int c = t; c < 512; c += 256) {
        int r   = c >> 2;
        int seg = c & 3;
        const void* ga = g + (size_t)r * D_DIM + seg * 8;
        unsigned loff = lbase + (unsigned)((r * LDT + seg * 8) * 2);
        asm volatile("global_load_async_to_lds_b128 %0, %1, off"
                     :: "v"(loff), "v"(ga) : "memory");
    }
}

// ---------------------------------------------------------------------------
// Kernel 3: bf16x3 split GEMM  out[m,n] = sum_k X[m,k]*W[n,k] + bias[n]
// Block tile 128x128, 8 waves (2x4), each wave 64x32 -> 4x2 WMMA tiles.
// Double-buffered LDS, async global->LDS tile copies overlapped with WMMA.
// ---------------------------------------------------------------------------
__global__ void __launch_bounds__(256)
gemm_bf16x3_kernel(const unsigned short* __restrict__ Xhi,
                   const unsigned short* __restrict__ Xlo,
                   const unsigned short* __restrict__ Whi,
                   const unsigned short* __restrict__ Wlo,
                   const float* __restrict__ bias,
                   float* __restrict__ out, int O) {
    // [stage][tile: Ah, Al, Bh, Bl][128*LDT halfs]  => 80 KB total
    __shared__ __align__(16) unsigned short sbuf[2][4][TILE_HALFS];

    const int t    = threadIdx.x;
    const int lane = t & 31;
    const int wave = t >> 5;          // 0..7
    const int wm   = wave >> 2;       // 0..1
    const int wn   = wave & 3;        // 0..3
    const int m0   = wm * 64;
    const int n0   = wn * 32;

    const size_t rowbase = (size_t)blockIdx.y * TILE_M;
    const size_t colbase = (size_t)blockIdx.x * TILE_N;

    const unsigned short* gAh = Xhi + rowbase * D_DIM;
    const unsigned short* gAl = Xlo + rowbase * D_DIM;
    const unsigned short* gBh = Whi + colbase * D_DIM;
    const unsigned short* gBl = Wlo + colbase * D_DIM;

    v8f acc[4][2];
    #pragma unroll
    for (int i = 0; i < 4; ++i)
        #pragma unroll
        for (int j = 0; j < 2; ++j)
            acc[i][j] = (v8f)(0.0f);

    auto issue_stage = [&](int kt, int buf) {
        const int koff = kt * TILE_K;
        async_tile_load(gAh + koff, lds_off(&sbuf[buf][0][0]), t);
        async_tile_load(gAl + koff, lds_off(&sbuf[buf][1][0]), t);
        async_tile_load(gBh + koff, lds_off(&sbuf[buf][2][0]), t);
        async_tile_load(gBl + koff, lds_off(&sbuf[buf][3][0]), t);
    };

    const int ksteps = D_DIM / TILE_K;   // 128
    issue_stage(0, 0);                    // prologue: 8 async b128 per thread

    for (int kt = 0; kt < ksteps; ++kt) {
        const int cur = kt & 1;
        if (kt + 1 < ksteps) {
            issue_stage(kt + 1, cur ^ 1); // overlap next stage with compute
            WAIT_ASYNC(8);                // first 8 (stage kt) complete, in order
        } else {
            WAIT_ASYNC(0);
        }
        __syncthreads();                  // stage kt visible to all waves

        const __bf16* sAh = (const __bf16*)sbuf[cur][0];
        const __bf16* sAl = (const __bf16*)sbuf[cur][1];
        const __bf16* sBh = (const __bf16*)sbuf[cur][2];
        const __bf16* sBl = (const __bf16*)sbuf[cur][3];

        v16bf bh[2], bl[2];
        #pragma unroll
        for (int j = 0; j < 2; ++j) {
            bh[j] = load_b_frag(sBh + (n0 + j * 16) * LDT);
            bl[j] = load_b_frag(sBl + (n0 + j * 16) * LDT);
        }

        #pragma unroll
        for (int i = 0; i < 4; ++i) {
            v16bf ah = load_a_frag(sAh + (m0 + i * 16) * LDT);
            v16bf al = load_a_frag(sAl + (m0 + i * 16) * LDT);
            #pragma unroll
            for (int j = 0; j < 2; ++j) {
                acc[i][j] = __builtin_amdgcn_wmma_f32_16x16x32_bf16(
                    false, ah, false, bh[j], (short)0, acc[i][j], false, false);
                acc[i][j] = __builtin_amdgcn_wmma_f32_16x16x32_bf16(
                    false, ah, false, bl[j], (short)0, acc[i][j], false, false);
                acc[i][j] = __builtin_amdgcn_wmma_f32_16x16x32_bf16(
                    false, al, false, bh[j], (short)0, acc[i][j], false, false);
            }
        }
        __syncthreads();   // done reading buf[cur] before stage kt+2 overwrites
    }

    // epilogue: C/D layout -> global + bias
    const int h = lane >> 4;
    const int nl = lane & 15;
    #pragma unroll
    for (int i = 0; i < 4; ++i) {
        #pragma unroll
        for (int j = 0; j < 2; ++j) {
            const size_t ng = colbase + n0 + j * 16 + nl;
            const float bv = bias[ng];
            #pragma unroll
            for (int r = 0; r < 8; ++r) {
                const size_t mg = rowbase + m0 + i * 16 + r + 8 * h;
                out[mg * (size_t)O + ng] = acc[i][j][r] + bv;
            }
        }
    }
}

// ---------------------------------------------------------------------------
// Host-side launcher
// ---------------------------------------------------------------------------
extern "C" void kernel_launch(void* const* d_in, const int* in_sizes, int n_in,
                              void* d_out, int out_size, void* d_ws, size_t ws_size,
                              hipStream_t stream) {
    const float* hidden = (const float*)d_in[0];   // [M, 4096] f32 (M = B*S)
    const float* weight = (const float*)d_in[1];   // [4096, 4096] f32
    const float* bias   = (const float*)d_in[2];   // [4096] f32
    float* out = (float*)d_out;

    const int D = D_DIM;
    const int M = in_sizes[0] / D;   // 8192
    const int O = in_sizes[2];       // 4096

    unsigned short* Xhi = (unsigned short*)d_ws;
    unsigned short* Xlo = Xhi + (size_t)M * D;
    unsigned short* Whi = Xlo + (size_t)M * D;
    unsigned short* Wlo = Whi + (size_t)O * D;

    fwht_split_kernel<<<M, 256, 0, stream>>>(hidden, Xhi, Xlo);

    int wblocks = (int)(((size_t)O * D) / (4 * 256));
    weight_split_kernel<<<wblocks, 256, 0, stream>>>(weight, Whi, Wlo);

    dim3 grid(O / TILE_N, M / TILE_M);   // (32, 64)
    gemm_bf16x3_kernel<<<grid, 256, 0, stream>>>(Xhi, Xlo, Whi, Wlo, bias, out, O);
}
